// TransitionSimple_GNN_80745385164943
// MI455X (gfx1250) — compile-verified, matrix operand order checked by reading, not run on hardware
//
#include <hip/hip_runtime.h>
#include <hip/hip_bf16.h>

typedef __attribute__((ext_vector_type(16))) _Float16 v16h;
typedef __attribute__((ext_vector_type(8)))  _Float16 v8h;
typedef __attribute__((ext_vector_type(8)))  float    v8f;

#define BK   8192      // B*K rows
#define F    64
#define H    128
#define KDIM 32        // edges per node (K in python)
#define IBLK 8         // dest-nodes i per edge-kernel block (hd/W2 reuse)

__device__ __forceinline__ v16h cat16(v8h lo, v8h hi) {
    return __builtin_shufflevector(lo, hi,
        0, 1, 2, 3, 4, 5, 6, 7, 8, 9, 10, 11, 12, 13, 14, 15);
}
__device__ __forceinline__ v8f wmma_f16(v16h a, v16h b, v8f c) {
    return __builtin_amdgcn_wmma_f32_16x16x32_f16(
        false, a, false, b, (short)0, c, false, false);
}

// ---------------------------------------------------------------------------
// Prep: swizzle all weights to f16 WMMA B-fragment order.
// frag[(nt*NKT+kt)*32 + lane][e] = B[c][16*nt + (lane&15)],
//   c = (e&8?16:0) + (e&7) + 8*(lane>>4) + 32*kt
// Regions: w1s 16384 | w2s 16384 | nw1s 24576 | nw2s 8192   (65536 total)
// ---------------------------------------------------------------------------
__global__ __launch_bounds__(256)
void prep_weights_kernel(const float* __restrict__ eW1,
                         const float* __restrict__ eW2,
                         const float* __restrict__ nW1,
                         const float* __restrict__ nW2,
                         _Float16* __restrict__ w1s,
                         _Float16* __restrict__ w2s,
                         _Float16* __restrict__ nw1s,
                         _Float16* __restrict__ nw2s) {
    const int idx  = blockIdx.x * 256 + threadIdx.x;   // 0..65535
    const int e    = idx & 15;
    const int lane = (idx >> 4) & 31;
    const int half = lane >> 4;
    const int m    = lane & 15;
    const int cb   = ((e & 8) ? 16 : 0) + (e & 7) + 8 * half;

    if (idx < 16384) {                       // w1s: B=[eW1[:64]|eW1[64:]] K=64,N=256
        const int local = idx;
        const int chunk = local >> 9;        // nt*2 + kt
        const int kt = chunk & 1, nt = chunk >> 1;
        const int c = cb + 32 * kt;          // 0..63
        const int n = 16 * nt + m;           // 0..255
        const float v = (n < H) ? eW1[c * H + n] : eW1[(F + c) * H + (n - H)];
        w1s[local] = (_Float16)v;
    } else if (idx < 32768) {                // w2s: eW2 K=128,N=128
        const int local = idx - 16384;
        const int chunk = local >> 9;
        const int kt = chunk & 3, nt = chunk >> 2;
        const int c = cb + 32 * kt;
        const int n = 16 * nt + m;
        w2s[local] = (_Float16)eW2[c * H + n];
    } else if (idx < 57344) {                // nw1s: nW1 K=192,N=128
        const int local = idx - 32768;
        const int chunk = local >> 9;        // 0..47
        const int kt = chunk % 6, nt = chunk / 6;
        const int c = cb + 32 * kt;          // 0..191
        const int n = 16 * nt + m;           // 0..127
        nw1s[local] = (_Float16)nW1[c * H + n];
    } else {                                 // nw2s: nW2 K=128,N=64
        const int local = idx - 57344;
        const int chunk = local >> 9;        // 0..15
        const int kt = chunk & 3, nt = chunk >> 2;
        const int c = cb + 32 * kt;          // 0..127
        const int n = 16 * nt + m;           // 0..63
        nw2s[local] = (_Float16)nW2[c * F + n];
    }
}

// ---------------------------------------------------------------------------
// Stage 1: hs|hd = x @ [eW1[:64] | eW1[64:]]  via WMMA.
// 256 blocks x 256 thr; block handles 32 rows: [32x64] @ [64x256].
// ---------------------------------------------------------------------------
__global__ __launch_bounds__(256)
void hs_hd_wmma_kernel(const float* __restrict__ x,
                       const float* __restrict__ eb1,
                       const _Float16* __restrict__ w1s,
                       float* __restrict__ hs,
                       float* __restrict__ hd) {
    const int blk  = blockIdx.x;
    const int tid  = threadIdx.x;
    const int w    = tid >> 5;
    const int lane = tid & 31;
    const int half = lane >> 4;
    const int m    = lane & 15;

    __shared__ __align__(32) _Float16 sX16[KDIM * F];   // 4 KB
    {
        const int j = tid >> 3, c0 = (tid & 7) * 8;
        const float* src = x + ((size_t)blk * KDIM + j) * F + c0;
        v8h p;
#pragma unroll
        for (int c = 0; c < 8; ++c) p[c] = (_Float16)src[c];
        *(v8h*)&sX16[j * F + c0] = p;
    }
    __syncthreads();

    const v16h* bf = (const v16h*)w1s;
#pragma unroll
    for (int nti = 0; nti < 2; ++nti) {
        const int nt = 2 * w + nti;
        v16h bfr[2];
#pragma unroll
        for (int kt = 0; kt < 2; ++kt) bfr[kt] = bf[(nt * 2 + kt) * 32 + lane];

        const int n = 16 * nt + m;
        float bias = 0.f;
        float* dst;
        int nn;
        if (nt < 8) { bias = eb1[n]; dst = hs; nn = n; }
        else        { dst = hd; nn = n - H; }

#pragma unroll
        for (int mt = 0; mt < 2; ++mt) {
            v8f acc = {};
            const int j = m + 16 * mt;
#pragma unroll
            for (int kt = 0; kt < 2; ++kt) {
                const int cb = 8 * half + 32 * kt;
                v8h lo = *(const v8h*)&sX16[j * F + cb];
                v8h hi = *(const v8h*)&sX16[j * F + cb + 16];
                acc = wmma_f16(cat16(lo, hi), bfr[kt], acc);
            }
#pragma unroll
            for (int r = 0; r < 8; ++r) {
                const int R = blk * KDIM + r + 8 * half + 16 * mt;
                dst[(size_t)R * H + nn] = acc[r] + bias;
            }
        }
    }
}

// ---------------------------------------------------------------------------
// Stage 2: block = (b, i0..i0+IBLK-1). hd and W2 fragments loaded once,
// reused for IBLK edge GEMMs: e = relu(hs_i + hd_j) @ eW2 + eb2 ;
// LayerNorm_H ; agg[g] = eg*sum_j norm + 32*ebt.
// ---------------------------------------------------------------------------
__global__ __launch_bounds__(256)
void edge_kernel(const float* __restrict__ eb2,
                 const float* __restrict__ eg,
                 const float* __restrict__ ebt,
                 const float* __restrict__ hs,
                 const float* __restrict__ hd,
                 const _Float16* __restrict__ w2s,
                 float* __restrict__ agg) {
    const int nb   = KDIM / IBLK;          // blocks per batch
    const int b    = blockIdx.x / nb;
    const int i0   = (blockIdx.x % nb) * IBLK;
    const int tid  = threadIdx.x;
    const int w    = tid >> 5;
    const int lane = tid & 31;
    const int half = lane >> 4;
    const int m    = lane & 15;

    __shared__ __align__(16) float    sHd[KDIM * H];    // 16 KB, persistent
    __shared__ __align__(16) float    sEo[KDIM * H];    // 16 KB, e output
    __shared__ __align__(32) _Float16 sH16[KDIM * H];   //  8 KB, per-i A matrix
    __shared__ __align__(16) float    sHs[IBLK * H];    //  4 KB, IBLK hs rows
    __shared__ float sPartS[KDIM][8];
    __shared__ float sPartQ[KDIM][8];
    __shared__ float sMu[KDIM];
    __shared__ float sRs[KDIM];

    {
        const uint4* src = (const uint4*)(hd + (size_t)b * KDIM * H);
        uint4*       dst = (uint4*)sHd;
        for (int idx = tid; idx < (KDIM * H) / 4; idx += 256) dst[idx] = src[idx];
    }
    {   // IBLK consecutive hs rows = IBLK*H floats = 256 uint4
        const uint4* src = (const uint4*)(hs + ((size_t)b * KDIM + i0) * H);
        ((uint4*)sHs)[tid] = src[tid];
    }

    // B fragments: one coalesced load per wave, live in VGPRs all kernel
    const v16h* bfp = (const v16h*)w2s;
    v16h bfr[4];
#pragma unroll
    for (int kt = 0; kt < 4; ++kt) bfr[kt] = bfp[(w * 4 + kt) * 32 + lane];
    __syncthreads();

    for (int ii = 0; ii < IBLK; ++ii) {
        // ---- stage h16[j][c] = f16(relu(hs_i[c] + hd[j][c])) -------------
        {
            const int j  = tid >> 3;
            const int c0 = (tid & 7) * 16;
            v8h p0, p1;
#pragma unroll
            for (int c = 0; c < 8; ++c) {
                float v = sHs[ii * H + c0 + c] + sHd[j * H + c0 + c];
                p0[c] = (_Float16)(v > 0.f ? v : 0.f);
            }
#pragma unroll
            for (int c = 0; c < 8; ++c) {
                float v = sHs[ii * H + c0 + 8 + c] + sHd[j * H + c0 + 8 + c];
                p1[c] = (_Float16)(v > 0.f ? v : 0.f);
            }
            *(v8h*)&sH16[j * H + c0]     = p0;
            *(v8h*)&sH16[j * H + c0 + 8] = p1;
        }
        __syncthreads();

        // ---- [32x128] @ [128x128] WMMA, scatter e (+eb2) -----------------
#pragma unroll
        for (int mt = 0; mt < 2; ++mt) {
            v8f acc = {};
            const int j = m + 16 * mt;
#pragma unroll
            for (int kt = 0; kt < 4; ++kt) {
                const int cb = 8 * half + 32 * kt;
                v8h lo = *(const v8h*)&sH16[j * H + cb];
                v8h hi = *(const v8h*)&sH16[j * H + cb + 16];
                acc = wmma_f16(cat16(lo, hi), bfr[kt], acc);
            }
            const int g = w * 16 + m;
#pragma unroll
            for (int r = 0; r < 8; ++r) {
                const int j2 = r + 8 * half + 16 * mt;
                sEo[j2 * H + g] = acc[r] + eb2[g];
            }
        }
        __syncthreads();

        // ---- LayerNorm stats per edge row j ------------------------------
        {
            const int j = tid >> 3, seg = tid & 7;
            const float* rowp = &sEo[j * H + seg * 16];
            float s = 0.f, q = 0.f;
#pragma unroll
            for (int c = 0; c < 16; ++c) { float v = rowp[c]; s += v; q += v * v; }
            sPartS[j][seg] = s;
            sPartQ[j][seg] = q;
        }
        __syncthreads();
        if (tid < KDIM) {
            float s = 0.f, q = 0.f;
#pragma unroll
            for (int seg = 0; seg < 8; ++seg) { s += sPartS[tid][seg]; q += sPartQ[tid][seg]; }
            const float mu  = s * (1.f / 128.f);
            const float var = q * (1.f / 128.f) - mu * mu;
            sMu[tid] = mu;
            sRs[tid] = rsqrtf(var + 1e-5f);
        }
        __syncthreads();

        // ---- agg[g] = eg*sum_j norm + 32*ebt -> global -------------------
        if (tid < H) {
            float s = 0.f;
            for (int j = 0; j < KDIM; ++j)
                s += (sEo[j * H + tid] - sMu[j]) * sRs[j];
            agg[((size_t)b * KDIM + i0 + ii) * H + tid] =
                eg[tid] * s + (float)KDIM * ebt[tid];
        }
        __syncthreads();   // protect sH16/sEo for next iteration
    }
}

// ---------------------------------------------------------------------------
// Stage 3: node MLP, batched 32 rows/block via WMMA.
//   n = relu([x|agg] @ nW1 + nb1)  : [32x192]@[192x128]
//   o = n @ nW2 + nb2              : [32x128]@[128x64]
//   out = LayerNorm_F(o)*ng + nbt
// ---------------------------------------------------------------------------
__global__ __launch_bounds__(256)
void node_kernel(const float* __restrict__ x,
                 const float* __restrict__ agg,
                 const float* __restrict__ nb1,
                 const float* __restrict__ nb2,
                 const float* __restrict__ ng,
                 const float* __restrict__ nbt,
                 const _Float16* __restrict__ nw1s,
                 const _Float16* __restrict__ nw2s,
                 float* __restrict__ out) {
    const int blk  = blockIdx.x;
    const int tid  = threadIdx.x;
    const int w    = tid >> 5;
    const int lane = tid & 31;
    const int half = lane >> 4;
    const int m    = lane & 15;

    __shared__ __align__(32) _Float16 sZ16[KDIM * 192];  // 12 KB
    __shared__ __align__(32) _Float16 sN16[KDIM * H];    //  8 KB
    __shared__ __align__(16) float    sO[KDIM * F];      //  8 KB
    __shared__ float sPartS[KDIM][8];
    __shared__ float sPartQ[KDIM][8];
    __shared__ float sMu[KDIM];
    __shared__ float sRs[KDIM];

    // stage z = [x | agg] as f16
    {
        const int j = tid >> 3, s8 = (tid & 7) * 8;      // x part: 64 ch
        const float* src = x + ((size_t)blk * KDIM + j) * F + s8;
        v8h p;
#pragma unroll
        for (int c = 0; c < 8; ++c) p[c] = (_Float16)src[c];
        *(v8h*)&sZ16[j * 192 + s8] = p;
    }
#pragma unroll
    for (int t2 = 0; t2 < 2; ++t2) {                     // agg part: 128 ch
        const int task = tid + 256 * t2;
        const int j = task >> 4, s8 = (task & 15) * 8;
        const float* src = agg + ((size_t)blk * KDIM + j) * H + s8;
        v8h p;
#pragma unroll
        for (int c = 0; c < 8; ++c) p[c] = (_Float16)src[c];
        *(v8h*)&sZ16[j * 192 + F + s8] = p;
    }
    __syncthreads();

    // GEMM1: wave w owns N-tile w (g = 16w..16w+15), K = 192 (6 chunks)
    {
        const v16h* bf = (const v16h*)nw1s;
        v16h bfr[6];
#pragma unroll
        for (int kt = 0; kt < 6; ++kt) bfr[kt] = bf[(w * 6 + kt) * 32 + lane];
        const int g = 16 * w + m;
        const float bias = nb1[g];
#pragma unroll
        for (int mt = 0; mt < 2; ++mt) {
            v8f acc = {};
            const int j = m + 16 * mt;
#pragma unroll
            for (int kt = 0; kt < 6; ++kt) {
                const int cb = 8 * half + 32 * kt;
                v8h lo = *(const v8h*)&sZ16[j * 192 + cb];
                v8h hi = *(const v8h*)&sZ16[j * 192 + cb + 16];
                acc = wmma_f16(cat16(lo, hi), bfr[kt], acc);
            }
#pragma unroll
            for (int r = 0; r < 8; ++r) {
                const int j2 = r + 8 * half + 16 * mt;
                float v = acc[r] + bias;
                sN16[j2 * H + g] = (_Float16)(v > 0.f ? v : 0.f);
            }
        }
    }
    __syncthreads();

    // GEMM2: wave w -> nt = w&3, mt = w>>2 ; K = 128 (4 chunks)
    {
        const v16h* bf = (const v16h*)nw2s;
        const int nt = w & 3, mt = w >> 2;
        const int n  = 16 * nt + m;
        const float bias = nb2[n];
        v8f acc = {};
        const int j = m + 16 * mt;
#pragma unroll
        for (int kt = 0; kt < 4; ++kt) {
            const int cb = 8 * half + 32 * kt;
            v8h lo = *(const v8h*)&sN16[j * H + cb];
            v8h hi = *(const v8h*)&sN16[j * H + cb + 16];
            acc = wmma_f16(cat16(lo, hi), bf[(nt * 4 + kt) * 32 + lane], acc);
        }
#pragma unroll
        for (int r = 0; r < 8; ++r) {
            const int j2 = r + 8 * half + 16 * mt;
            sO[j2 * F + n] = acc[r] + bias;
        }
    }
    __syncthreads();

    // LayerNorm over F=64 per row
    {
        const int j = tid >> 3, seg = tid & 7;
        const float* rowp = &sO[j * F + seg * 8];
        float s = 0.f, q = 0.f;
#pragma unroll
        for (int c = 0; c < 8; ++c) { float v = rowp[c]; s += v; q += v * v; }
        sPartS[j][seg] = s;
        sPartQ[j][seg] = q;
    }
    __syncthreads();
    if (tid < KDIM) {
        float s = 0.f, q = 0.f;
#pragma unroll
        for (int seg = 0; seg < 8; ++seg) { s += sPartS[tid][seg]; q += sPartQ[tid][seg]; }
        const float mu  = s * (1.f / (float)F);
        const float var = q * (1.f / (float)F) - mu * mu;
        sMu[tid] = mu;
        sRs[tid] = rsqrtf(var + 1e-5f);
    }
    __syncthreads();
    {
        const int j = tid >> 3, c0 = (tid & 7) * 8;
        const size_t R = (size_t)blk * KDIM + j;
#pragma unroll
        for (int c = 0; c < 8; ++c) {
            const int cc = c0 + c;
            out[R * F + cc] =
                (sO[j * F + cc] - sMu[j]) * sRs[j] * ng[cc] + nbt[cc];
        }
    }
}

// ---------------------------------------------------------------------------
extern "C" void kernel_launch(void* const* d_in, const int* in_sizes, int n_in,
                              void* d_out, int out_size, void* d_ws, size_t ws_size,
                              hipStream_t stream) {
    (void)in_sizes; (void)n_in; (void)out_size; (void)ws_size;

    const float* x   = (const float*)d_in[0];
    const float* eW1 = (const float*)d_in[1];
    const float* eb1 = (const float*)d_in[2];
    const float* eW2 = (const float*)d_in[3];
    const float* eb2 = (const float*)d_in[4];
    const float* eg  = (const float*)d_in[5];
    const float* ebt = (const float*)d_in[6];
    const float* nW1 = (const float*)d_in[7];
    const float* nb1 = (const float*)d_in[8];
    const float* nW2 = (const float*)d_in[9];
    const float* nb2 = (const float*)d_in[10];
    const float* ng  = (const float*)d_in[11];
    const float* nbt = (const float*)d_in[12];
    float* out = (float*)d_out;

    // ws: hs | hd | agg (f32, 4MB each) then f16 swizzled weights
    float*    ws_hs   = (float*)d_ws;
    float*    ws_hd   = ws_hs  + (size_t)BK * H;
    float*    ws_agg  = ws_hd  + (size_t)BK * H;
    _Float16* ws_w1s  = (_Float16*)(ws_agg + (size_t)BK * H);
    _Float16* ws_w2s  = ws_w1s  + 16384;
    _Float16* ws_nw1s = ws_w2s  + 16384;
    _Float16* ws_nw2s = ws_nw1s + 24576;

    prep_weights_kernel<<<256, 256, 0, stream>>>(
        eW1, eW2, nW1, nW2, ws_w1s, ws_w2s, ws_nw1s, ws_nw2s);
    hs_hd_wmma_kernel<<<BK / KDIM, 256, 0, stream>>>(x, eb1, ws_w1s, ws_hs, ws_hd);
    edge_kernel<<<BK / IBLK, 256, 0, stream>>>(eb2, eg, ebt, ws_hs, ws_hd, ws_w2s, ws_agg);
    node_kernel<<<BK / KDIM, 256, 0, stream>>>(
        x, ws_agg, nb1, nb2, ng, nbt, ws_nw1s, ws_nw2s, out);
}
